// ContextualAttention_42460046689012
// MI455X (gfx1250) — matrix operand order repaired and checked by reading, not compiled, not run on hardware
//
#include <hip/hip_runtime.h>
#include <hip/hip_bf16.h>

// CDNA5 wave32 / WMMA bf16 implementation of contextual attention.
// Workspace layout (bytes), total ~247.6 MB:
//   [0,              37.75M) fg_mat   bf16 (B,1024,1152)   -- later reused as G2 (f32, col-major 1152x1024)
//   [37.75M,         75.50M) bsn_mat  bf16                  -- (part of G2 alias)
//   [75.50M,        113.25M) bi_mat   bf16 (B,1024,1152)
//   [113.25M,       113.31M) valid    f32  (B,1024)
//   [113.31M,       180.42M) St       f32  (B,1024,1024)  (S transposed: St[q][p])
//   [180.42M,       247.53M) T1       f32  (fuse ping)    -- later reused as Sbf (bf16 scores)

typedef __attribute__((ext_vector_type(16))) __bf16 v16bf;
typedef __attribute__((ext_vector_type(8)))  __bf16 v8bf;
typedef __attribute__((ext_vector_type(8)))  float  v8f;

#define B_  16
#define C_  128
#define H_  64
#define L_  1024
#define KP  1152   // C*9

// ---------------------------------------------------------------- prep
// One WG per (b,p): 128 threads = channels. Builds im2col rows (k = j*128+c,
// c contiguous -> coalesced bf16 writes), L2-normalizes bs, computes valid.
__global__ __launch_bounds__(128)
void prep_kernel(const float* __restrict__ f_o, const float* __restrict__ b_o,
                 const float* __restrict__ mask_o,
                 __bf16* __restrict__ fg, __bf16* __restrict__ bsn,
                 __bf16* __restrict__ bim, float* __restrict__ valid)
{
  const int bp = blockIdx.x;
  const int b = bp >> 10, p = bp & 1023;
  const int py = p >> 5, px = p & 31;
  const int c = threadIdx.x;
  const float* fb = f_o + ((size_t)b * C_ + c) * (H_ * H_);
  const float* bb = b_o + ((size_t)b * C_ + c) * (H_ * H_);
  float fv[9], bv[9], biv[9];
  float ss = 0.f;
#pragma unroll
  for (int j = 0; j < 9; ++j) {
    const int dy = j / 3 - 1, dx = j % 3 - 1;
    const int yy = py + dy, xx = px + dx;                  // downsampled grid
    const bool in = ((unsigned)yy < 32u) && ((unsigned)xx < 32u);
    const float fvv = in ? fb[(yy * 2) * H_ + xx * 2] : 0.f;
    const float bvv = in ? bb[(yy * 2) * H_ + xx * 2] : 0.f;
    const int oy = 2 * py + dy, ox = 2 * px + dx;          // stride-2 patches of b_o
    const bool in2 = ((unsigned)oy < 64u) && ((unsigned)ox < 64u);
    const float bivv = in2 ? bb[oy * H_ + ox] : 0.f;
    fv[j] = fvv; bv[j] = bvv; biv[j] = bivv;
    ss += bvv * bvv;
  }
  // sum of squares across all 128 channels (wave32 shuffle + LDS)
#pragma unroll
  for (int o = 16; o > 0; o >>= 1) ss += __shfl_xor(ss, o, 32);
  __shared__ float red[4];
  if ((threadIdx.x & 31) == 0) red[threadIdx.x >> 5] = ss;
  __syncthreads();
  const float tot = red[0] + red[1] + red[2] + red[3];
  const float inv = 1.f / fmaxf(sqrtf(tot), 1e-4f);        // EPS = 1e-4
  if (threadIdx.x == 0) {
    const float* mb = mask_o + (size_t)b * (H_ * H_);
    float s = 0.f;
#pragma unroll
    for (int j = 0; j < 9; ++j) {
      const int yy = py + j / 3 - 1, xx = px + j % 3 - 1;
      if (((unsigned)yy < 32u) && ((unsigned)xx < 32u)) s += mb[(yy * 2) * H_ + xx * 2];
    }
    valid[bp] = (s == 0.f) ? 1.f : 0.f;                    // mean==0 <=> sum==0 (mask>=0)
  }
  const size_t base = (size_t)bp * KP;
#pragma unroll
  for (int j = 0; j < 9; ++j) {
    fg [base + j * C_ + c] = (__bf16)fv[j];
    bsn[base + j * C_ + c] = (__bf16)(bv[j] * inv);
    bim[base + j * C_ + c] = (__bf16)biv[j];
  }
}

// ---------------------------------------------------------------- GEMM
// D[m][n] = sum_k A[m][k] * B'[n][k].  A: MxK row-major (lda=K, bf16).
// BNN=false: Bm is NxK row-major (NT form).  BNN=true: Bm is KxN row-major,
// transposed into LDS during staging.  COLOUT: store D column-major (ldD=M)
// through an LDS-staged epilogue so downstream reads are contiguous in m.
//
// 128 threads = 4 waves; WG tile 64x128; wave tile 32x64 (2x4 16x16 frags,
// 8 v_wmma_f32_16x16x32_bf16 per k-step); K-step 32; double-buffered LDS with
// a one-barrier-per-step software pipeline (step s+1 global loads issued
// before step s WMMAs so XDL work covers L2/HBM latency).
__device__ __forceinline__ v16bf cat16(v8bf lo, v8bf hi) {
  return __builtin_shufflevector(lo, hi, 0, 1, 2, 3, 4, 5, 6, 7,
                                 8, 9, 10, 11, 12, 13, 14, 15);
}

template <bool BNN, bool COLOUT>
__global__ __launch_bounds__(128)
void gemm_wmma(const __bf16* __restrict__ A, const __bf16* __restrict__ Bm,
               float* __restrict__ D, int K, int ldB, int ldD,
               size_t bsA, size_t bsB, size_t bsD)
{
  // Row stride 40 halves = 80 B: 16B-aligned for ds_load_b128, and the
  // per-row bank rotation (20 dwords) keeps fragment gathers conflict-light.
  __shared__ __attribute__((aligned(16))) __bf16 sA[2][64 * 40];
  __shared__ __attribute__((aligned(16))) __bf16 sB[2][128 * 40];
  const int tid = threadIdx.x;
  const int m0 = blockIdx.x * 64, n0 = blockIdx.y * 128;
  A  += bsA * blockIdx.z;
  Bm += bsB * blockIdx.z;
  D  += bsD * blockIdx.z;
  const int lane = tid & 31, wave = tid >> 5;
  const int wy = wave >> 1, wx = wave & 1;
  const int half = lane >> 4, l16 = lane & 15;
  v8f acc[8] = {};

  // Staging thread mappings (all global reads are 16B-aligned, coalesced).
  const int ar = tid >> 1, aseg = (tid & 1) * 16;           // A: 2 threads/row
  const __bf16* aptr = A + (size_t)(m0 + ar) * K + aseg;
  const int br = tid;                                       // B NT: 1 thread/row
  const __bf16* bptrNT = Bm + (size_t)(n0 + br) * ldB;
  const int kr = tid >> 2, nseg = (tid & 3) * 32;           // B NN: 4 threads/row
  const __bf16* bptrNN = Bm + (size_t)kr * ldB + n0 + nseg;

  v8bf ra0, ra1, rb0, rb1, rb2, rb3;
  auto loadRegs = [&](int k0) {
    const __bf16* pa = aptr + k0;
    ra0 = *(const v8bf*)pa;        ra1 = *(const v8bf*)(pa + 8);
    if constexpr (!BNN) {
      const __bf16* pb = bptrNT + k0;
      rb0 = *(const v8bf*)pb;        rb1 = *(const v8bf*)(pb + 8);
      rb2 = *(const v8bf*)(pb + 16); rb3 = *(const v8bf*)(pb + 24);
    } else {
      const __bf16* pb = bptrNN + (size_t)k0 * ldB;
      rb0 = *(const v8bf*)pb;        rb1 = *(const v8bf*)(pb + 8);
      rb2 = *(const v8bf*)(pb + 16); rb3 = *(const v8bf*)(pb + 24);
    }
  };
  auto storeRegs = [&](int buf) {
    *(v8bf*)&sA[buf][ar * 40 + aseg]     = ra0;
    *(v8bf*)&sA[buf][ar * 40 + aseg + 8] = ra1;
    if constexpr (!BNN) {
      *(v8bf*)&sB[buf][br * 40]      = rb0;
      *(v8bf*)&sB[buf][br * 40 + 8]  = rb1;
      *(v8bf*)&sB[buf][br * 40 + 16] = rb2;
      *(v8bf*)&sB[buf][br * 40 + 24] = rb3;
    } else {                        // transpose 32 halves into n-major rows
#pragma unroll
      for (int e = 0; e < 8; ++e) {
        sB[buf][(nseg + e)      * 40 + kr] = rb0[e];
        sB[buf][(nseg + 8 + e)  * 40 + kr] = rb1[e];
        sB[buf][(nseg + 16 + e) * 40 + kr] = rb2[e];
        sB[buf][(nseg + 24 + e) * 40 + kr] = rb3[e];
      }
    }
  };

  loadRegs(0);
  storeRegs(0);
  __syncthreads();

  const int nsteps = K >> 5;
  for (int s = 0; s < nsteps; ++s) {
    const int cur = s & 1;
    if (s + 1 < nsteps) loadRegs((s + 1) << 5);   // prefetch next tile

    const __bf16* bufA = sA[cur];
    const __bf16* bufB = sB[cur];
    // Fragment gathers in contiguous k order, k = 16*half + s (two 16B LDS loads each).
    v16bf af[2], bfr[4];
#pragma unroll
    for (int mf = 0; mf < 2; ++mf) {
      // A: lanes 0-15 hold K 0..15; lanes 16-31 hold K 16..31
      const __bf16* pa = &bufA[(wy * 32 + mf * 16 + l16) * 40 + half * 16];
      af[mf] = cat16(*(const v8bf*)pa, *(const v8bf*)(pa + 8));
    }
#pragma unroll
    for (int nf = 0; nf < 4; ++nf) {
      // B: column N = lane&15; lanes 0-15 K 0..15, lanes 16-31 K 16..31
      const __bf16* pb = &bufB[(wx * 64 + nf * 16 + l16) * 40 + half * 16];
      bfr[nf] = cat16(*(const v8bf*)pb, *(const v8bf*)(pb + 8));
    }
#pragma unroll
    for (int mf = 0; mf < 2; ++mf)
#pragma unroll
      for (int nf = 0; nf < 4; ++nf)
        acc[mf * 4 + nf] = __builtin_amdgcn_wmma_f32_16x16x32_bf16(
            false, af[mf], false, bfr[nf], (short)0, acc[mf * 4 + nf],
            false, false);

    if (s + 1 < nsteps) storeRegs(cur ^ 1);       // fill the other buffer
    __syncthreads();
  }

  if constexpr (!COLOUT) {
    // Direct row-major stores: each accum VGPR is a contiguous 16-lane 64B run.
#pragma unroll
    for (int mf = 0; mf < 2; ++mf)
#pragma unroll
      for (int nf = 0; nf < 4; ++nf) {
        const int gm = m0 + wy * 32 + mf * 16 + half * 8;
        const int gn = n0 + wx * 64 + nf * 16 + l16;
        float* dst = D + (size_t)gm * ldD + gn;
#pragma unroll
        for (int j = 0; j < 8; ++j) dst[(size_t)j * ldD] = acc[mf * 4 + nf][j];
      }
  } else {
    // Column-major store through LDS (pad 129 -> (m+n)%64 bank rotation).
    __shared__ __attribute__((aligned(16))) float sD[64 * 129];
#pragma unroll
    for (int mf = 0; mf < 2; ++mf)
#pragma unroll
      for (int nf = 0; nf < 4; ++nf) {
        const int ml = wy * 32 + mf * 16 + half * 8;
        const int nl = wx * 64 + nf * 16 + l16;
#pragma unroll
        for (int j = 0; j < 8; ++j) sD[(ml + j) * 129 + nl] = acc[mf * 4 + nf][j];
      }
    __syncthreads();
    const int n = tid;                            // full-line 256B global runs
    float* dst = D + (size_t)(n0 + n) * ldD + m0;
#pragma unroll
    for (int i = 0; i < 64; i += 4) {
      float4 v = make_float4(sD[(i)     * 129 + n], sD[(i + 1) * 129 + n],
                             sD[(i + 2) * 129 + n], sD[(i + 3) * 129 + n]);
      *(float4*)(dst + i) = v;
    }
  }
}

// ---------------------------------------------------------------- fuse
// diag_fuse: out[i][j] = sum_{d=-1..1, bounds} in[i+d][j+d] (flat 1024 grid).
__global__ __launch_bounds__(256)
void fuse1_kernel(const float* __restrict__ in, float* __restrict__ out)
{
  const size_t idx = (size_t)blockIdx.x * 256 + threadIdx.x;
  const size_t mat = idx >> 20;
  const int ij = (int)(idx & 1048575);
  const int i = ij >> 10, j = ij & 1023;
  const float* S = in + (mat << 20);
  float s = S[ij];
  if (i > 0 && j > 0)       s += S[ij - 1025];
  if (i < 1023 && j < 1023) s += S[ij + 1025];
  out[idx] = s;
}

__device__ __forceinline__ int swp(int v) { return ((v & 31) << 5) | (v >> 5); }

// Second diag_fuse in (x,y)-swapped index space (transpose-commuting, so it
// applies to St identically): out[i][j] = sum_d in[swp(swp(i)+d)][swp(swp(j)+d)].
__global__ __launch_bounds__(256)
void fuse2_kernel(const float* __restrict__ in, float* __restrict__ out)
{
  const size_t idx = (size_t)blockIdx.x * 256 + threadIdx.x;
  const size_t mat = idx >> 20;
  const int ij = (int)(idx & 1048575);
  const int i = ij >> 10, j = ij & 1023;
  const float* S = in + (mat << 20);
  const int a = swp(i), bq = swp(j);
  float s = S[ij];
  if (a > 0 && bq > 0)       s += S[(size_t)swp(a - 1) * 1024 + swp(bq - 1)];
  if (a < 1023 && bq < 1023) s += S[(size_t)swp(a + 1) * 1024 + swp(bq + 1)];
  out[idx] = s;
}

// ---------------------------------------------------------------- softmax
// One WG per (b,q) row of St (contiguous in p). Applies valid mask before and
// after softmax over p, SS=10; emits bf16 scores for the second WMMA GEMM.
__global__ __launch_bounds__(256)
void softmax_kernel(const float* __restrict__ St, const float* __restrict__ valid,
                    __bf16* __restrict__ outbf)
{
  const int row = blockIdx.x;  // b*1024 + q
  const float* r = St + (size_t)row * 1024;
  const float* vl = valid + (size_t)(row >> 10) * 1024;
  __shared__ float sm[8];
  float v[4];
  float mx = -3.0e38f;
#pragma unroll
  for (int i = 0; i < 4; ++i) {
    const int p = threadIdx.x + i * 256;
    v[i] = r[p] * vl[p] * 10.0f;
    mx = fmaxf(mx, v[i]);
  }
#pragma unroll
  for (int o = 16; o > 0; o >>= 1) mx = fmaxf(mx, __shfl_xor(mx, o, 32));
  if ((threadIdx.x & 31) == 0) sm[threadIdx.x >> 5] = mx;
  __syncthreads();
  mx = sm[threadIdx.x & 7];
#pragma unroll
  for (int o = 4; o > 0; o >>= 1) mx = fmaxf(mx, __shfl_xor(mx, o, 32));
  __syncthreads();
  float sum = 0.f;
#pragma unroll
  for (int i = 0; i < 4; ++i) { v[i] = __expf(v[i] - mx); sum += v[i]; }
#pragma unroll
  for (int o = 16; o > 0; o >>= 1) sum += __shfl_xor(sum, o, 32);
  if ((threadIdx.x & 31) == 0) sm[threadIdx.x >> 5] = sum;
  __syncthreads();
  sum = sm[threadIdx.x & 7];
#pragma unroll
  for (int o = 4; o > 0; o >>= 1) sum += __shfl_xor(sum, o, 32);
  const float invs = 1.f / sum;
#pragma unroll
  for (int i = 0; i < 4; ++i) {
    const int p = threadIdx.x + i * 256;
    outbf[(size_t)row * 1024 + p] = (__bf16)(v[i] * invs * vl[p]);
  }
}

// ---------------------------------------------------------------- fold
// col2im of the dilated transposed conv: out[b][c][oy][ox] = 0.25 * sum over
// (jy,jx) with oy = 2*qy + jy - 1 (and same in x) of G2[(jy*3+jx)*128+c][q].
// G2 is column-major (q contiguous) so lanes along ox read contiguous runs.
__global__ __launch_bounds__(256)
void fold_kernel(const float* __restrict__ G2, float* __restrict__ out)
{
  const int b = blockIdx.x, c = blockIdx.y;
  const int oy = blockIdx.z * 4 + threadIdx.y;
  const int ox = threadIdx.x;
  const float* g = G2 + (size_t)b * ((size_t)KP * L_);
  float s = 0.f;
#pragma unroll
  for (int jy = 0; jy < 3; ++jy) {
    const int ty = oy + 1 - jy;
    if (ty & 1) continue;
    const int qy = ty >> 1;
    if ((unsigned)qy >= 32u) continue;
#pragma unroll
    for (int jx = 0; jx < 3; ++jx) {
      const int tx = ox + 1 - jx;
      if (tx & 1) continue;
      const int qx = tx >> 1;
      if ((unsigned)qx >= 32u) continue;
      s += g[(size_t)((jy * 3 + jx) * C_ + c) * L_ + qy * 32 + qx];
    }
  }
  out[(((size_t)b * C_ + c) * H_ + oy) * H_ + ox] = 0.25f * s;
}

// ---------------------------------------------------------------- launch
extern "C" void kernel_launch(void* const* d_in, const int* /*in_sizes*/, int /*n_in*/,
                              void* d_out, int /*out_size*/, void* d_ws, size_t /*ws_size*/,
                              hipStream_t stream)
{
  const float* f_o = (const float*)d_in[0];
  const float* b_o = (const float*)d_in[1];
  const float* m_o = (const float*)d_in[2];
  float* out = (float*)d_out;
  char* ws = (char*)d_ws;

  const size_t szPatch = (size_t)B_ * L_ * KP * sizeof(__bf16);  // 37.75 MB
  __bf16* fg    = (__bf16*)(ws);
  __bf16* bsn   = (__bf16*)(ws + szPatch);
  __bf16* bim   = (__bf16*)(ws + 2 * szPatch);
  float*  valid = (float*)(ws + 3 * szPatch);
  char*   p4    = ws + 3 * szPatch + (size_t)B_ * L_ * sizeof(float);
  float*  St    = (float*)(p4);                                   // 64 MB
  float*  T1    = (float*)(p4 + (size_t)B_ * L_ * L_ * sizeof(float));  // 64 MB
  __bf16* Sbf   = (__bf16*)T1;       // alias: T1 is dead after fuse2
  float*  G2    = (float*)ws;        // alias: fg+bsn (75.5 MB) dead after GEMM1

  prep_kernel<<<dim3(B_ * L_), dim3(128), 0, stream>>>(f_o, b_o, m_o,
                                                       fg, bsn, bim, valid);

  // GEMM1 (NT): St[q][p] = sum_k fg[q][k] * bsn[p][k]   (K=1152)
  gemm_wmma<false, false><<<dim3(16, 8, B_), dim3(128), 0, stream>>>(
      fg, bsn, St, KP, KP, L_,
      (size_t)L_ * KP, (size_t)L_ * KP, (size_t)L_ * L_);

  const int NEL = (B_ * L_ * L_) / 256;
  fuse1_kernel<<<dim3(NEL), dim3(256), 0, stream>>>(St, T1);
  fuse2_kernel<<<dim3(NEL), dim3(256), 0, stream>>>(T1, St);
  softmax_kernel<<<dim3(B_ * L_), dim3(256), 0, stream>>>(St, valid, Sbf);

  // GEMM2 (NN-B, col-major out): G2[k'][q] = sum_p Sbf[q][p] * bim[p][k']  (K=1024)
  gemm_wmma<true, true><<<dim3(16, 9, B_), dim3(128), 0, stream>>>(
      Sbf, bim, G2, L_, KP, L_,
      (size_t)L_ * L_, (size_t)L_ * KP, (size_t)L_ * KP);

  fold_kernel<<<dim3(B_, C_, 16), dim3(64, 4), 0, stream>>>(G2, out);
}
